// Attention_10368051052857
// MI455X (gfx1250) — matrix operand order repaired and checked by reading, not compiled, not run on hardware
//
#include <hip/hip_runtime.h>
#include <hip/hip_bf16.h>

// ---------------------------------------------------------------------------
// Self-attention over a 64x64 feature map, bf16 WMMA pipeline for gfx1250.
//   K1: QKV projection GEMM (f32 in -> bf16 Q*scale, K, V^T)
//   K2: flash attention, software-pipelined TDM staging (K/V DMA overlaps
//       WMMA compute via TENSORcnt), online softmax
//   K3: output projection + bias + residual (TDM-staged O tile, f32 out)
// ---------------------------------------------------------------------------

typedef __bf16 bf16;
typedef __attribute__((ext_vector_type(16))) __bf16 v16bf;
typedef __attribute__((ext_vector_type(8)))  float  v8f;
typedef __attribute__((ext_vector_type(4)))  unsigned int v4u;
typedef __attribute__((ext_vector_type(8)))  int v8i;
typedef __attribute__((ext_vector_type(4)))  int v4i;

#define NTOK  4096   // H*W
#define CDIM  256
#define INNER 512
#define QSCALE 0.125f

__device__ __forceinline__ v8f vzero() {
  v8f z = {0.f, 0.f, 0.f, 0.f, 0.f, 0.f, 0.f, 0.f};
  return z;
}

__device__ __forceinline__ unsigned lds_off(const void* p) {
  // Low 32 bits of the flat address of an LDS object = LDS byte offset.
  return (unsigned)(unsigned long long)p;
}

// ---------------------------------------------------------------------------
// Tensor Data Mover: 2-D tile load global -> LDS (D# descriptor, ISA ch.8).
// rows x cols elements of 2-byte data, row stride in elements; contiguous in
// LDS at lds_byte_off. Issue from one wave; completion via TENSORcnt.
// ---------------------------------------------------------------------------
__device__ __forceinline__ void tdm_load_2d(unsigned lds_byte_off, const void* gptr,
                                            unsigned rows, unsigned cols,
                                            unsigned row_stride_elems) {
  unsigned long long ga = (unsigned long long)gptr;
  v4u g0;
  g0[0] = 1u;                                   // count=1, user desc, no gather
  g0[1] = lds_byte_off;                         // D#.lds_addr  (bits 63:32)
  g0[2] = (unsigned)ga;                         // global_addr[31:0]
  g0[3] = ((unsigned)(ga >> 32) & 0x01FFFFFFu)  // global_addr[56:32]
          | 0x80000000u;                        // type=2 ("image") bits 127:126
  v8i g1;
  g1[0] = (int)(1u << 16);                      // data_size=1 (2 bytes), mask=0
  g1[1] = (int)(cols << 16);                    // tensor_dim0[15:0]  @ 63:48
  g1[2] = (int)((cols >> 16) | (rows << 16));   // tensor_dim0 hi, tensor_dim1 lo
  g1[3] = (int)(((rows >> 16) & 0xFFFFu)        // tensor_dim1 hi
                | (cols << 16));                // tile_dim0 @ 127:112
  g1[4] = (int)(rows & 0xFFFFu);                // tile_dim1 ; tile_dim2=0
  g1[5] = (int)row_stride_elems;                // tensor_dim0_stride[31:0]
  g1[6] = 0;
  g1[7] = 0;
  v4i z4 = {0, 0, 0, 0};
#if __clang_major__ >= 23
  v8i z8 = {0, 0, 0, 0, 0, 0, 0, 0};
  __builtin_amdgcn_tensor_load_to_lds(g0, g1, z4, z4, z8, 0);
#else
  __builtin_amdgcn_tensor_load_to_lds(g0, g1, z4, z4, 0);
#endif
}

// A fragment: 16(M) x 32(K) bf16, element (m,k) at p[m*rs + k*cs].
__device__ __forceinline__ v16bf load_a_frag(const bf16* p, int rs, int cs) {
  const int lane = threadIdx.x & 31;
  const int m  = lane & 15;
  const int kb = (lane >> 4) << 4;   // 0 or 16
  v16bf a;
#pragma unroll
  for (int i = 0; i < 8; ++i) {
    a[i]     = p[m * rs + (kb + i) * cs];
    a[i + 8] = p[m * rs + (kb + 8 + i) * cs];
  }
  return a;
}

// B fragment: 32(K) x 16(N) bf16, element (k,n) at p[k*rs + n*cs].
__device__ __forceinline__ v16bf load_b_frag(const bf16* p, int rs, int cs) {
  const int lane = threadIdx.x & 31;
  const int n  = lane & 15;
  const int kb = (lane >> 4) << 4;   // 0 or 16
  v16bf b;
#pragma unroll
  for (int i = 0; i < 16; ++i)
    b[i] = p[(kb + i) * rs + n * cs];
  return b;
}

__device__ __forceinline__ v8f wmma_bf16(v16bf a, v16bf b, v8f c) {
  return __builtin_amdgcn_wmma_f32_16x16x32_bf16(
      /*neg_a=*/false, a, /*neg_b=*/false, b,
      /*c_mod=*/(short)0, c, /*reuse_a=*/false, /*reuse_b=*/false);
}

// ---------------------------------------------------------------------------
// Kernel 1: qkv[b,n,o] = sum_c x[b,c,n] * w_qkv[o,c]   (o in [0,1536))
// Q,K stored [b][n][d]; V stored transposed [b][d][n] so the attention
// kernel's PV B-fragments are k-contiguous in LDS.
// grid = (B*N/64, 1536/64)
// ---------------------------------------------------------------------------
__global__ __launch_bounds__(256) void qkv_kernel(
    const float* __restrict__ x, const float* __restrict__ w_qkv,
    bf16* __restrict__ Q, bf16* __restrict__ K, bf16* __restrict__ Vt) {
  __shared__ __align__(16) bf16 As[64 * 40];   // x tile  [n_local][c_local]
  __shared__ __align__(16) bf16 Bs[64 * 40];   // w tile  [o_local][c_local]

  const int tid  = threadIdx.x;
  const int wave = tid >> 5;
  const int lane = tid & 31;
  const int b    = blockIdx.x >> 6;
  const int n0   = (blockIdx.x & 63) << 6;
  const int o0   = blockIdx.y << 6;
  const float* xb = x + (size_t)b * CDIM * NTOK;

  const int mrow = (wave >> 1) << 4;   // 0,16,32,48
  const int ncol = (wave & 1) << 5;    // 0,32
  v8f acc0 = vzero(), acc1 = vzero();

  for (int k0 = 0; k0 < CDIM; k0 += 32) {
    // Stage A transposed: x[b][k0+i][n0+j] -> As[j][i]  (f32 -> bf16)
#pragma unroll
    for (int s = 0; s < 2; ++s) {
      int c = tid * 2 + s;                   // 512 float4 chunks
      int i = c >> 4, j = (c & 15) << 2;
      const float4 f = *(const float4*)&xb[(size_t)(k0 + i) * NTOK + n0 + j];
      As[(j + 0) * 40 + i] = (bf16)f.x;
      As[(j + 1) * 40 + i] = (bf16)f.y;
      As[(j + 2) * 40 + i] = (bf16)f.z;
      As[(j + 3) * 40 + i] = (bf16)f.w;
    }
    // Stage B: w_qkv[o0+j][k0+i] -> Bs[j][i]
#pragma unroll
    for (int s = 0; s < 2; ++s) {
      int c = tid * 2 + s;                   // 512 float4 chunks
      int j = c >> 3, i = (c & 7) << 2;
      const float4 f = *(const float4*)&w_qkv[(size_t)(o0 + j) * CDIM + k0 + i];
      bf16* d = &Bs[j * 40 + i];
      d[0] = (bf16)f.x; d[1] = (bf16)f.y; d[2] = (bf16)f.z; d[3] = (bf16)f.w;
    }
    __syncthreads();
    v16bf a  = load_a_frag(&As[mrow * 40], /*rs=*/40, /*cs=*/1);
    v16bf b0 = load_b_frag(&Bs[(ncol + 0) * 40], /*rs=*/1, /*cs=*/40);
    v16bf b1 = load_b_frag(&Bs[(ncol + 16) * 40], 1, 40);
    acc0 = wmma_bf16(a, b0, acc0);
    acc1 = wmma_bf16(a, b1, acc1);
    __syncthreads();
  }

  const int colL = lane & 15;
  const int rb   = (lane >> 4) << 3;
  const size_t nb = (size_t)b * NTOK;

#pragma unroll
  for (int t = 0; t < 2; ++t) {
    const v8f acc = t ? acc1 : acc0;
    const int o = o0 + ncol + t * 16 + colL;
    const int which = o >> 9;          // 0=Q 1=K 2=V
    const int d = o & 511;
    if (which < 2) {
      bf16* dst = which ? K : Q;
      const float scl = which ? 1.0f : QSCALE;
#pragma unroll
      for (int e = 0; e < 8; ++e) {
        int n = n0 + mrow + rb + e;
        dst[(nb + n) * INNER + d] = (bf16)(acc[e] * scl);
      }
    } else {
      bf16* dst = Vt + ((size_t)b * INNER + d) * NTOK;   // [d][n]
#pragma unroll
      for (int e = 0; e < 8; ++e)
        dst[n0 + mrow + rb + e] = (bf16)acc[e];
    }
  }
}

// ---------------------------------------------------------------------------
// Kernel 2: flash attention. 64 queries per WG, KV blocks of 64.
// TDM pipelining (wave 0 issues; TENSORcnt is in-order per wave):
//   top of iter j : wait cnt==0  -> K[j] resident
//                   issue V[j]           (DMA overlaps S = Q K^T WMMAs)
//   after S store : issue K[j+1]         (DMA overlaps softmax + PV)
//   before PV     : wait cnt<=1  -> V[j] resident
// grid = B * N/64
// ---------------------------------------------------------------------------
#define SS_LD 65
#define PS_LD 68

__global__ __launch_bounds__(256) void attn_kernel(
    const bf16* __restrict__ Q, const bf16* __restrict__ K,
    const bf16* __restrict__ Vt, bf16* __restrict__ O) {
  __shared__ __align__(16) bf16  Qs[64 * INNER];    // 64 KB (TDM, once)
  __shared__ __align__(16) bf16  Ks[64 * INNER];    // 64 KB (TDM, [j][d])
  __shared__ __align__(16) bf16  Vts[INNER * 64];   // 64 KB (TDM, [d][j])
  __shared__ __align__(16) float Ss[64 * SS_LD];
  __shared__ __align__(16) bf16  Ps[64 * PS_LD];
  __shared__ float rowM[64], rowL[64], rowA[64];

  const int tid  = threadIdx.x;
  const int wave = tid >> 5;
  const int lane = tid & 31;
  const int b    = blockIdx.x >> 6;
  const int n0   = (blockIdx.x & 63) << 6;

  const bf16* Qg  = Q  + ((size_t)b * NTOK + n0) * INNER;
  const bf16* Kg  = K  + (size_t)b * NTOK * INNER;
  const bf16* Vtg = Vt + (size_t)b * INNER * NTOK;   // [d][n]

  if (wave == 0) {
    tdm_load_2d(lds_off(Qs), Qg, 64, INNER, INNER);
    tdm_load_2d(lds_off(Ks), Kg, 64, INNER, INNER);
  }
  if (tid < 64) { rowM[tid] = -1e30f; rowL[tid] = 0.f; }

  v8f oacc[4][4];
#pragma unroll
  for (int i = 0; i < 4; ++i)
#pragma unroll
    for (int j = 0; j < 4; ++j) oacc[i][j] = vzero();

  const int srow = (wave >> 1) << 4;
  const int scol = (wave & 1) << 5;
  const int rb   = (lane >> 4) << 3;
  const int colL = lane & 31 & 15;
  const unsigned ks_lds  = lds_off(Ks);
  const unsigned vts_lds = lds_off(Vts);

  for (int j0 = 0; j0 < NTOK; j0 += 64) {
    const bool last = (j0 + 64 >= NTOK);
    // K[j] (and, on first iter, Q) resident for wave 0; publish to all waves.
    if (wave == 0) __builtin_amdgcn_s_wait_tensorcnt((short)0);
    __syncthreads();
    // Kick off V[j] DMA; it streams in while we do the S WMMAs below.
    if (wave == 0)
      tdm_load_2d(vts_lds, Vtg + j0, INNER, 64, NTOK);

    // --- S = Q K^T (full d = 512) ---
    v8f s0 = vzero(), s1 = vzero();
    for (int kk = 0; kk < INNER; kk += 32) {
      v16bf a   = load_a_frag(&Qs[srow * INNER + kk], INNER, 1);
      v16bf bb0 = load_b_frag(&Ks[(scol + 0) * INNER + kk], 1, INNER);
      v16bf bb1 = load_b_frag(&Ks[(scol + 16) * INNER + kk], 1, INNER);
      s0 = wmma_bf16(a, bb0, s0);
      s1 = wmma_bf16(a, bb1, s1);
    }
#pragma unroll
    for (int e = 0; e < 8; ++e) {
      Ss[(srow + rb + e) * SS_LD + scol + colL]      = s0[e];
      Ss[(srow + rb + e) * SS_LD + scol + 16 + colL] = s1[e];
    }
    // warm L2 for the K block after next
    if (j0 + 128 < NTOK)
      __builtin_prefetch(&Kg[(size_t)(j0 + 128 + (tid & 63)) * INNER], 0, 1);
    __syncthreads();

    // All S reads of Ks done: kick off K[j+1] DMA (overlaps softmax + PV).
    if (wave == 0 && !last)
      tdm_load_2d(ks_lds, Kg + (size_t)(j0 + 64) * INNER, 64, INNER, INNER);

    // --- online softmax (one row per thread, tid<64) ---
    if (tid < 64) {
      const float* sr = &Ss[tid * SS_LD];
      float mold = rowM[tid];
      float mnew = mold;
      for (int j = 0; j < 64; ++j) mnew = fmaxf(mnew, sr[j]);
      float alpha = __expf(mold - mnew);
      float lsum = 0.f;
      bf16* pr = &Ps[tid * PS_LD];
      for (int j = 0; j < 64; ++j) {
        float p = __expf(sr[j] - mnew);
        lsum += p;
        pr[j] = (bf16)p;
      }
      rowM[tid] = mnew;
      rowL[tid] = rowL[tid] * alpha + lsum;
      rowA[tid] = alpha;
    }
    __syncthreads();

    // --- rescale O accumulators by alpha(row) ---
#pragma unroll
    for (int rt = 0; rt < 4; ++rt)
#pragma unroll
      for (int e = 0; e < 8; ++e) {
        float al = rowA[rt * 16 + rb + e];
#pragma unroll
        for (int ct = 0; ct < 4; ++ct) oacc[rt][ct][e] *= al;
      }

    // V[j] resident once the older of {V[j], K[j+1]} has retired (in-order).
    if (wave == 0) {
      if (last) __builtin_amdgcn_s_wait_tensorcnt((short)0);
      else      __builtin_amdgcn_s_wait_tensorcnt((short)1);
    }
    __syncthreads();

    // --- O += P @ V  (V^T tile: element (k=j, n=d) at Vts[n*64 + k]) ---
    for (int kk = 0; kk < 64; kk += 32) {
      v16bf bb[4];
#pragma unroll
      for (int ct = 0; ct < 4; ++ct)
        bb[ct] = load_b_frag(&Vts[(wave * 64 + ct * 16) * 64 + kk], 1, 64);
#pragma unroll
      for (int rt = 0; rt < 4; ++rt) {
        v16bf a = load_a_frag(&Ps[(rt * 16) * PS_LD + kk], PS_LD, 1);
#pragma unroll
        for (int ct = 0; ct < 4; ++ct)
          oacc[rt][ct] = wmma_bf16(a, bb[ct], oacc[rt][ct]);
      }
    }
    __syncthreads();
  }

  // --- epilogue: O / l -> bf16 [B][N][512] ---
#pragma unroll
  for (int rt = 0; rt < 4; ++rt)
#pragma unroll
    for (int e = 0; e < 8; ++e) {
      int r = rt * 16 + rb + e;
      float inv = 1.0f / rowL[r];
#pragma unroll
      for (int ct = 0; ct < 4; ++ct) {
        int d = wave * 64 + ct * 16 + colL;
        O[((size_t)b * NTOK + n0 + r) * INNER + d] = (bf16)(oacc[rt][ct][e] * inv);
      }
    }
}

// ---------------------------------------------------------------------------
// Kernel 3: out[b,c,n] = x[b,c,n] + b_out[c] + sum_d O[b,n,d] * w_out[c,d]
// Whole 64x512 bf16 O tile TDM-staged once; w_out chunks converted per k-step.
// grid = (B*N/64, C/64)
// ---------------------------------------------------------------------------
__global__ __launch_bounds__(256) void proj_kernel(
    const bf16* __restrict__ Obf, const float* __restrict__ w_out,
    const float* __restrict__ b_out, const float* __restrict__ x,
    float* __restrict__ out) {
  __shared__ __align__(16) bf16 Osh[64 * INNER];   // 64 KB (TDM)
  __shared__ __align__(16) bf16 As[64 * 40];       // w_out tile [c_l][d_l]

  const int tid  = threadIdx.x;
  const int wave = tid >> 5;
  const int lane = tid & 31;
  const int b    = blockIdx.x >> 6;
  const int n0   = (blockIdx.x & 63) << 6;
  const int c0   = blockIdx.y << 6;

  const int mrow = (wave >> 1) << 4;   // c offset in tile
  const int ncol = (wave & 1) << 5;    // n offset in tile
  v8f acc0 = vzero(), acc1 = vzero();

  const bf16* Og = Obf + ((size_t)b * NTOK + n0) * INNER;
  if (wave == 0) {
    tdm_load_2d(lds_off(Osh), Og, 64, INNER, INNER);
    __builtin_amdgcn_s_wait_tensorcnt((short)0);
  }

  for (int k0 = 0; k0 < INNER; k0 += 32) {
#pragma unroll
    for (int s = 0; s < 2; ++s) {
      int c = tid * 2 + s;
      int j = c >> 3, i = (c & 7) << 2;
      const float4 f = *(const float4*)&w_out[(size_t)(c0 + j) * INNER + k0 + i];
      bf16* d = &As[j * 40 + i];
      d[0] = (bf16)f.x; d[1] = (bf16)f.y; d[2] = (bf16)f.z; d[3] = (bf16)f.w;
    }
    __syncthreads();
    v16bf a   = load_a_frag(&As[mrow * 40], /*rs=*/40, /*cs=*/1);
    v16bf bb0 = load_b_frag(&Osh[(ncol + 0) * INNER + k0], /*rs=*/1, /*cs=*/INNER);
    v16bf bb1 = load_b_frag(&Osh[(ncol + 16) * INNER + k0], 1, INNER);
    acc0 = wmma_bf16(a, bb0, acc0);
    acc1 = wmma_bf16(a, bb1, acc1);
    __syncthreads();
  }

  const int colL = lane & 15;
  const int rb   = (lane >> 4) << 3;
#pragma unroll
  for (int e = 0; e < 8; ++e) {
    int c = c0 + mrow + rb + e;
    float bo = b_out[c];
    size_t base = ((size_t)b * CDIM + c) * NTOK + n0;
    int na = ncol + colL, nb2 = ncol + 16 + colL;
    out[base + na]  = acc0[e] + bo + x[base + na];
    out[base + nb2] = acc1[e] + bo + x[base + nb2];
  }
}

// ---------------------------------------------------------------------------
extern "C" void kernel_launch(void* const* d_in, const int* in_sizes, int n_in,
                              void* d_out, int out_size, void* d_ws, size_t ws_size,
                              hipStream_t stream) {
  (void)in_sizes; (void)n_in; (void)out_size; (void)ws_size;
  const float* x     = (const float*)d_in[0];
  const float* w_qkv = (const float*)d_in[1];
  const float* w_out = (const float*)d_in[2];
  const float* b_out = (const float*)d_in[3];
  float* out = (float*)d_out;

  const size_t elems = (size_t)4 * NTOK * INNER;   // 8 Mi elements each
  bf16* Q  = (bf16*)d_ws;
  bf16* K  = Q + elems;
  bf16* Vt = K + elems;                            // transposed [b][d][n]
  bf16* Ob = Vt + elems;                           // total 64 MiB of ws

  dim3 g1(4 * NTOK / 64, (3 * INNER) / 64);        // (256, 24)
  qkv_kernel<<<g1, 256, 0, stream>>>(x, w_qkv, Q, K, Vt);

  attn_kernel<<<4 * NTOK / 64, 256, 0, stream>>>(Q, K, Vt, Ob);

  dim3 g3(4 * NTOK / 64, CDIM / 64);               // (256, 4)
  proj_kernel<<<g3, 256, 0, stream>>>(Ob, w_out, b_out, x, out);
}
